// Contra_balsc_ours_30889404793117
// MI455X (gfx1250) — matrix operand order repaired and checked by reading, not compiled, not run on hardware
//
#include <hip/hip_runtime.h>
#include <hip/hip_bf16.h>

// ---------------- problem constants ----------------
#define NB      4096                  // batch rows
#define NC      1000                  // classes / centers
#define NO      4096                  // ood rows
#define ND      512                   // feature dim
#define NTOT    (NB + NC + NO)        // 9192 columns
#define NPAD    9216                  // padded to 64-col strips (zero rows 9192..9215)
#define MTILE   128                   // rows per workgroup
#define NSTRIP  64                    // cols per strip
#define KCHUNK  64                    // K per LDS stage
#define YSPLIT  8
#define NSTRIPS_TOTAL (NPAD / NSTRIP)            // 144
#define STRIPS_PER_Y  (NSTRIPS_TOTAL / YSPLIT)   // 18
#define NSLOTS  (YSPLIT * 2)          // 2 n-groups per WG write separate slots
#define INVT    10.0f                 // 1/temperature; also the exact row max
#define LDA     72                    // padded LDS row stride (u16 elements)

typedef unsigned short u16;
typedef u16   v8u  __attribute__((ext_vector_type(8)));
typedef u16   v16u __attribute__((ext_vector_type(16)));
typedef __bf16 v16bf __attribute__((ext_vector_type(16)));
typedef float  v8f   __attribute__((ext_vector_type(8)));

static __device__ inline u16 f2bf(float f) {           // RNE f32 -> bf16 bits
    unsigned u = __float_as_uint(f);
    unsigned r = u + 0x7FFFu + ((u >> 16) & 1u);
    return (u16)(r >> 16);
}
static __device__ inline float bf2f(u16 h) {
    return __uint_as_float(((unsigned)h) << 16);
}

// A fragment: 16x32 bf16, lane = {m = lane&15, kh = lane>>4}
// element e -> k = (e<8 ? 8*kh + e : 16 + 8*kh + (e-8)); two 16B runs.
static __device__ inline v16bf ldfragA(const u16* p) {
    v8u a0 = *(const v8u*)(p);
    v8u a1 = *(const v8u*)(p + 16);
    v16u r = __builtin_shufflevector(a0, a1, 0,1,2,3,4,5,6,7,8,9,10,11,12,13,14,15);
    return __builtin_bit_cast(v16bf, r);
}
// B fragment: 32x16 bf16, lane = {n = lane&15, kh = lane>>4}, k = (e<8 ? 8*kh + e : 16 + 8*kh + (e-8))
static __device__ inline v16bf ldfragB(const u16* p) {
    v8u a0 = *(const v8u*)(p);
    v8u a1 = *(const v8u*)(p + 16);
    v16u r = __builtin_shufflevector(a0, a1, 0,1,2,3,4,5,6,7,8,9,10,11,12,13,14,15);
    return __builtin_bit_cast(v16bf, r);
}

// async global->LDS copy, 16B per lane, tracked by ASYNCcnt (gfx1250)
static __device__ inline void async_b128(unsigned lds_off, unsigned long long gaddr) {
    asm volatile("global_load_async_to_lds_b128 %0, %1, off"
                 :: "v"(lds_off), "v"(gaddr) : "memory");
}

// ---------------- phase 1: class counts + per-column tables ----------------
__global__ void k_init_counts(float* counts) {
    int i = blockIdx.x * blockDim.x + threadIdx.x;
    if (i < NC) counts[i] = 1.0f;     // each class center contributes one
}
__global__ void k_accum_counts(const int* targets, const int* pseudo, float* counts) {
    int i = blockIdx.x * blockDim.x + threadIdx.x;
    if (i < NB) atomicAdd(&counts[targets[i]], 1.0f);          // +1.0 adds are exact -> order independent
    else if (i < NB + NO) atomicAdd(&counts[pseudo[i - NB]], 1.0f);
}
__global__ void k_cols(const int* targets, const int* pseudo, const float* counts,
                       float* colw, int* label) {
    int j = blockIdx.x * blockDim.x + threadIdx.x;
    if (j >= NTOT) return;
    int lab, cls;
    if (j < NB)            { lab = targets[j];  cls = lab; }
    else if (j < NB + NC)  { lab = j - NB;      cls = lab; }
    else                   { lab = NC;          cls = pseudo[j - NB - NC]; }
    label[j] = lab;
    colw[j]  = counts[cls];
}

// ---------------- phase 2a: one-time f32 -> bf16 hi/lo split of features_all ----------------
__global__ void k_split(const float* __restrict__ centers1, const float* __restrict__ features,
                        const float* __restrict__ fo, u16* __restrict__ hi, u16* __restrict__ lo) {
    int idx = blockIdx.x * blockDim.x + threadIdx.x;     // 4 elements per thread
    long long e = (long long)idx * 4;
    if (e >= (long long)NPAD * ND) return;
    int row = (int)(e >> 9);
    int k   = (int)(e & 511);
    float4 v = make_float4(0.f, 0.f, 0.f, 0.f);
    const float* src = (row < NB)      ? features + (size_t)row * ND
                     : (row < NB + NC) ? centers1 + (size_t)(row - NB) * ND
                     : (row < NTOT)    ? fo + (size_t)(row - NB - NC) * ND
                                       : nullptr;
    if (src) v = *(const float4*)(src + k);
    float xs[4] = {v.x, v.y, v.z, v.w};
    u16 h[4], l[4];
#pragma unroll
    for (int q = 0; q < 4; ++q) {
        h[q] = f2bf(xs[q]);
        l[q] = f2bf(xs[q] - bf2f(h[q]));
    }
    unsigned h01 = (unsigned)h[0] | ((unsigned)h[1] << 16);
    unsigned h23 = (unsigned)h[2] | ((unsigned)h[3] << 16);
    unsigned l01 = (unsigned)l[0] | ((unsigned)l[1] << 16);
    unsigned l23 = (unsigned)l[2] | ((unsigned)l[3] << 16);
    *(uint2*)(hi + e) = make_uint2(h01, h23);
    *(uint2*)(lo + e) = make_uint2(l01, l23);
}

// ---------------- phase 2b: fused split-bf16 WMMA GEMM + masked reduction ----------------
__global__ __launch_bounds__(256)
void k_main(const u16* __restrict__ wsHi, const u16* __restrict__ wsLo,
            const int* __restrict__ targets,
            const float* __restrict__ colw, const int* __restrict__ label,
            float* __restrict__ Sarr, float* __restrict__ Parr, float* __restrict__ Carr) {
    // double-buffered LDS tiles (bf16 hi/lo), padded stride 72 -> 108 KB total
    __shared__ __align__(16) u16 AsH[2][MTILE * LDA];
    __shared__ __align__(16) u16 AsL[2][MTILE * LDA];
    __shared__ __align__(16) u16 BsH[2][NSTRIP * LDA];
    __shared__ __align__(16) u16 BsL[2][NSTRIP * LDA];
    __shared__ int tgt_lds[MTILE];

    const int t       = threadIdx.x;
    const int wave    = t >> 5;
    const int lane    = t & 31;
    const int nl      = lane & 15;
    const int kh      = lane >> 4;
    const int mg      = wave >> 1;     // M group 0..3 (32 rows each)
    const int ng      = wave & 1;      // N group 0..1 (32 cols each)
    const int rowBase = blockIdx.x * MTILE;

    if (t < MTILE) tgt_lds[t] = targets[rowBase + t];

    // per-lane running accumulators over strips: rows fixed per (mt, r)
    float sS[2][8], sP[2][8], sCnt[2][8];
#pragma unroll
    for (int mt = 0; mt < 2; ++mt)
#pragma unroll
        for (int r = 0; r < 8; ++r) { sS[mt][r] = 0.f; sP[mt][r] = 0.f; sCnt[mt][r] = 0.f; }

    for (int s = 0; s < STRIPS_PER_Y; ++s) {
        const int jBase = (blockIdx.y * STRIPS_PER_Y + s) * NSTRIP;
        v8f acc[2][2] = {};

        // ---- issue async copies for K-chunk 0 into buffer 0 ----
        // A: 128x64 -> 1024 x 16B (4/thread); B: 64x64 -> 512 x 16B (2/thread). x(hi,lo) = 12/thread.
        {
#pragma unroll
            for (int q = 0; q < 4; ++q) {
                int i = t + 256 * q, row = i >> 3, kq = (i & 7) << 3;
                unsigned long long goff = 2ull * ((unsigned long long)(rowBase + row) * ND + kq);
                unsigned loff = (unsigned)(row * LDA + kq) * 2u;
                async_b128((unsigned)(unsigned long long)(const void*)&AsH[0][0] + loff,
                           (unsigned long long)(const void*)wsHi + goff);
                async_b128((unsigned)(unsigned long long)(const void*)&AsL[0][0] + loff,
                           (unsigned long long)(const void*)wsLo + goff);
            }
#pragma unroll
            for (int q = 0; q < 2; ++q) {
                int i = t + 256 * q, col = i >> 3, kq = (i & 7) << 3;
                unsigned long long goff = 2ull * ((unsigned long long)(jBase + col) * ND + kq);
                unsigned loff = (unsigned)(col * LDA + kq) * 2u;
                async_b128((unsigned)(unsigned long long)(const void*)&BsH[0][0] + loff,
                           (unsigned long long)(const void*)wsHi + goff);
                async_b128((unsigned)(unsigned long long)(const void*)&BsL[0][0] + loff,
                           (unsigned long long)(const void*)wsLo + goff);
            }
        }

        for (int kc = 0; kc < ND / KCHUNK; ++kc) {
            const int cur = kc & 1;
            if (kc + 1 < ND / KCHUNK) {
                const int nxt = cur ^ 1, k0 = (kc + 1) * KCHUNK;
#pragma unroll
                for (int q = 0; q < 4; ++q) {
                    int i = t + 256 * q, row = i >> 3, kq = (i & 7) << 3;
                    unsigned long long goff = 2ull * ((unsigned long long)(rowBase + row) * ND + k0 + kq);
                    unsigned loff = (unsigned)(row * LDA + kq) * 2u;
                    async_b128((unsigned)(unsigned long long)(const void*)&AsH[nxt][0] + loff,
                               (unsigned long long)(const void*)wsHi + goff);
                    async_b128((unsigned)(unsigned long long)(const void*)&AsL[nxt][0] + loff,
                               (unsigned long long)(const void*)wsLo + goff);
                }
#pragma unroll
                for (int q = 0; q < 2; ++q) {
                    int i = t + 256 * q, col = i >> 3, kq = (i & 7) << 3;
                    unsigned long long goff = 2ull * ((unsigned long long)(jBase + col) * ND + k0 + kq);
                    unsigned loff = (unsigned)(col * LDA + kq) * 2u;
                    async_b128((unsigned)(unsigned long long)(const void*)&BsH[nxt][0] + loff,
                               (unsigned long long)(const void*)wsHi + goff);
                    async_b128((unsigned)(unsigned long long)(const void*)&BsL[nxt][0] + loff,
                               (unsigned long long)(const void*)wsLo + goff);
                }
                asm volatile("s_wait_asynccnt 12" ::: "memory");  // current chunk landed; next in flight
            } else {
                asm volatile("s_wait_asynccnt 0" ::: "memory");
            }
            __syncthreads();

            // ---- 2 K-sub-chunks of 32; 2x2 tiles; split product hi*hi + hi*lo + lo*hi ----
#pragma unroll
            for (int ks = 0; ks < KCHUNK; ks += 32) {
                v16bf aH[2], aL[2], bH[2], bL[2];
#pragma unroll
                for (int mt = 0; mt < 2; ++mt) {
                    const int aoff = (mg * 32 + mt * 16 + nl) * LDA + ks + 8 * kh;
                    aH[mt] = ldfragA(&AsH[cur][aoff]);
                    aL[mt] = ldfragA(&AsL[cur][aoff]);
                }
#pragma unroll
                for (int nt = 0; nt < 2; ++nt) {
                    const int boff = (ng * 32 + nt * 16 + nl) * LDA + ks + 8 * kh;
                    bH[nt] = ldfragB(&BsH[cur][boff]);
                    bL[nt] = ldfragB(&BsL[cur][boff]);
                }
#pragma unroll
                for (int mt = 0; mt < 2; ++mt)
#pragma unroll
                    for (int nt = 0; nt < 2; ++nt) {
                        acc[mt][nt] = __builtin_amdgcn_wmma_f32_16x16x32_bf16(false, aH[mt], false, bH[nt], (short)0, acc[mt][nt], false, false);
                        acc[mt][nt] = __builtin_amdgcn_wmma_f32_16x16x32_bf16(false, aH[mt], false, bL[nt], (short)0, acc[mt][nt], false, false);
                        acc[mt][nt] = __builtin_amdgcn_wmma_f32_16x16x32_bf16(false, aL[mt], false, bH[nt], (short)0, acc[mt][nt], false, false);
                    }
            }
            __syncthreads();   // all reads of this buffer done before it is refilled
        }

        // ---- epilogue: masked, weighted accumulation for the 2x2 tiles ----
#pragma unroll
        for (int nt = 0; nt < 2; ++nt) {
            const int j = jBase + ng * 32 + nt * 16 + nl;
            const bool jvalid = (j < NTOT);
            const float colwj = jvalid ? colw[j] : 1.0f;
            const int   labj  = jvalid ? label[j] : -1;
#pragma unroll
            for (int mt = 0; mt < 2; ++mt) {
#pragma unroll
                for (int r = 0; r < 8; ++r) {
                    const int m   = mg * 32 + mt * 16 + 8 * kh + r;  // C/D layout: M = r + 8*kh
                    const int row = rowBase + m;
                    const float l = acc[mt][nt][r] * INVT;           // logit
                    if (jvalid && (j != row)) {
                        const bool mk = (labj == tgt_lds[m]);
                        const float w = colwj - (mk ? 1.0f : 0.0f);
                        sS[mt][r] += __expf(l - INVT) / w;
                        if (mk) { sP[mt][r] += l; sCnt[mt][r] += 1.0f; }
                    }
                }
            }
        }
    }

    // ---- reduce across the 16 lanes sharing a row, store to this (y, ng) slot ----
#pragma unroll
    for (int mt = 0; mt < 2; ++mt)
#pragma unroll
        for (int r = 0; r < 8; ++r) {
            float vs = sS[mt][r], vp = sP[mt][r], vc = sCnt[mt][r];
#pragma unroll
            for (int d = 1; d < 16; d <<= 1) {
                vs += __shfl_xor(vs, d, 32);
                vp += __shfl_xor(vp, d, 32);
                vc += __shfl_xor(vc, d, 32);
            }
            if (nl == 0) {
                const int row  = rowBase + mg * 32 + mt * 16 + 8 * kh + r;
                const int slot = (blockIdx.y * 2 + ng) * NB + row;   // unique writer: plain store
                Sarr[slot] = vs; Parr[slot] = vp; Carr[slot] = vc;
            }
        }
}

// ---------------- phase 3: deterministic finalize ----------------
__global__ void k_finalize(const float* Sarr, const float* Parr, const float* Carr, float* out) {
    __shared__ float red[256];
    float local = 0.0f;
    for (int i = threadIdx.x; i < NB; i += 256) {
        float S = 0.0f, P = 0.0f, C = 0.0f;
#pragma unroll
        for (int y = 0; y < NSLOTS; ++y) {              // fixed order -> deterministic
            S += Sarr[y * NB + i];
            P += Parr[y * NB + i];
            C += Carr[y * NB + i];
        }
        local += P / C - INVT - logf(S);
    }
    red[threadIdx.x] = local;
    __syncthreads();
    for (int st = 128; st > 0; st >>= 1) {
        if (threadIdx.x < st) red[threadIdx.x] += red[threadIdx.x + st];
        __syncthreads();
    }
    if (threadIdx.x == 0) out[0] = -red[0] / (float)NB;
}

extern "C" void kernel_launch(void* const* d_in, const int* in_sizes, int n_in,
                              void* d_out, int out_size, void* d_ws, size_t ws_size,
                              hipStream_t stream) {
    const float* centers1 = (const float*)d_in[0];
    const float* features = (const float*)d_in[1];
    const int*   targets  = (const int*)d_in[2];
    const float* fo       = (const float*)d_in[3];
    const int*   pseudo   = (const int*)d_in[4];
    float* out = (float*)d_out;

    // workspace layout (bytes)
    const size_t SPLIT_BYTES = (size_t)NPAD * ND * 2;   // 9,437,184 each
    char* ws = (char*)d_ws;
    u16*   wsHi   = (u16*)  (ws + 0);
    u16*   wsLo   = (u16*)  (ws + SPLIT_BYTES);
    float* counts = (float*)(ws + 2 * SPLIT_BYTES);                   // 1000 f32
    float* colw   = (float*)(ws + 2 * SPLIT_BYTES + 4096);            // 9192 f32
    int*   label  = (int*)  (ws + 2 * SPLIT_BYTES + 4096 + 36864);    // 9192 i32
    float* Sarr   = (float*)(ws + 2 * SPLIT_BYTES + 4096 + 2 * 36864);
    float* Parr   = Sarr + (size_t)NSLOTS * NB;
    float* Carr   = Parr + (size_t)NSLOTS * NB;

    k_init_counts<<<(NC + 255) / 256, 256, 0, stream>>>(counts);
    k_accum_counts<<<(NB + NO + 255) / 256, 256, 0, stream>>>(targets, pseudo, counts);
    k_cols<<<(NTOT + 255) / 256, 256, 0, stream>>>(targets, pseudo, counts, colw, label);
    k_split<<<(NPAD * ND / 4 + 255) / 256, 256, 0, stream>>>(centers1, features, fo, wsHi, wsLo);

    dim3 grid(NB / MTILE, YSPLIT);   // 32 x 8 = 256 workgroups
    k_main<<<grid, 256, 0, stream>>>(wsHi, wsLo, targets, colw, label, Sarr, Parr, Carr);
    k_finalize<<<1, 256, 0, stream>>>(Sarr, Parr, Carr, out);
}
